// Simple_LSTM_45148696215894
// MI455X (gfx1250) — matrix operand order repaired and checked by reading, not compiled, not run on hardware
//
#include <hip/hip_runtime.h>

// ---------------------------------------------------------------------------
// Problem constants (from the reference)
// ---------------------------------------------------------------------------
#define VOCAB 32000
#define EMBED 256
#define HID   512
#define SEQ   128
#define BATCH 256
#define U1    256
#define U2    64
#define KTOT  (EMBED + HID)   // 768, fused [x_t | h] contraction dim
#define G4H   (4 * HID)       // 2048
#define KCH   128             // K-chunk per LDS stage
#define NSTAGE (KTOT / KCH)   // 6 stages
#define PANEL_BYTES (4 * 16 * KCH * 2)   // 4 gates x 16 cols x 128 k x bf16 = 16KB

// ---------------------------------------------------------------------------
// CDNA5 WMMA types (wave32, 16x16x32 bf16 -> f32)
// ---------------------------------------------------------------------------
typedef __attribute__((ext_vector_type(16))) __bf16 v16bf;
typedef __attribute__((ext_vector_type(8)))  __bf16 v8bf;
typedef __attribute__((ext_vector_type(8)))  float  v8f;

#define WMMA_BF16(a, b, c) \
    __builtin_amdgcn_wmma_f32_16x16x32_bf16(false, (a), false, (b), (short)0, (c), false, false)

// A-fragment loader: 16-bit A matrix 16x32 (MxK), row-major memory.
// Lane l: M = l&15, kgroup = l>>4. VGPR0-3 hold K = kg*8 + 0..7,
// VGPR4-7 hold K = 16 + kg*8 + 0..7  => two contiguous 16B chunks.
__device__ __forceinline__ v16bf load_a16(const __bf16* p) {
    v8bf lo = *(const v8bf*)(p);
    v8bf hi = *(const v8bf*)(p + 16);
    v16bf r;
#pragma unroll
    for (int i = 0; i < 8; ++i) { r[i] = lo[i]; r[8 + i] = hi[i]; }
    return r;
}

// Combine two 16B chunks already in LDS into a B fragment (ds_load_b128 x2).
__device__ __forceinline__ v16bf load_b16_lds(const __bf16* p) {
    v8bf lo = *(const v8bf*)(p);
    v8bf hi = *(const v8bf*)(p + 16);
    v16bf r;
#pragma unroll
    for (int i = 0; i < 8; ++i) { r[i] = lo[i]; r[8 + i] = hi[i]; }
    return r;
}

__device__ __forceinline__ float sigmoidf_dev(float x) {
    return 1.0f / (1.0f + __expf(-x));
}

// Raw 32-bit LDS byte offset of a generic pointer into a __shared__ object.
__device__ __forceinline__ unsigned lds_off(const void* p) {
    return (unsigned)(__SIZE_TYPE__)(const __attribute__((address_space(3))) char*)p;
}

// CDNA5 async copy: global -> LDS, 16B per lane, tracked by ASYNCcnt.
__device__ __forceinline__ void async_copy_b128(unsigned lds_byte_addr,
                                                const void* gaddr) {
    asm volatile("global_load_async_to_lds_b128 %0, %1, off"
                 :: "v"(lds_byte_addr), "v"(gaddr)
                 : "memory");
}

__device__ __forceinline__ void wait_asynccnt_le4() {
    asm volatile("s_wait_asynccnt 0x4" ::: "memory");
}

// ---------------------------------------------------------------------------
// Kernel 1: embedding gather + f32 -> bf16 cast.  x_bf16[b*SEQ+t][e]
// ---------------------------------------------------------------------------
__global__ void embed_cast_kernel(const int* __restrict__ inputs,
                                  const float* __restrict__ emb,
                                  unsigned short* __restrict__ xbf,
                                  int total) {
    int gid = blockIdx.x * blockDim.x + threadIdx.x;
    if (gid >= total) return;
    int row = gid >> 8;          // / EMBED
    int e   = gid & (EMBED - 1);
    int tok = inputs[row];
    float v = emb[(size_t)tok * EMBED + e];
    ((__bf16*)xbf)[gid] = (__bf16)v;
}

// ---------------------------------------------------------------------------
// Kernel 2: transpose + cast weights.  dst[n][doff + k] = (bf16) src[k][n]
// ---------------------------------------------------------------------------
__global__ void transpose_cast_kernel(const float* __restrict__ src,
                                      unsigned short* __restrict__ dst,
                                      int K, int N, int dstride, int doff,
                                      int total) {
    int gid = blockIdx.x * blockDim.x + threadIdx.x;
    if (gid >= total) return;
    int k = gid / N;
    int n = gid - k * N;
    ((__bf16*)dst)[(size_t)n * dstride + doff + k] = (__bf16)src[(size_t)k * N + n];
}

// ---------------------------------------------------------------------------
// Kernel 3: zero-init h0 (bf16) and c (f32)
// ---------------------------------------------------------------------------
__global__ void init_state_kernel(unsigned short* __restrict__ h0,
                                  float* __restrict__ c, int total) {
    int gid = blockIdx.x * blockDim.x + threadIdx.x;
    if (gid >= total) return;
    h0[gid] = 0;
    c[gid]  = 0.0f;
}

// ---------------------------------------------------------------------------
// Kernel 4: fused LSTM timestep with async-LDS weight staging.
//   z = [x_t | h_prev] @ [W;U]^T + b ; c = f*c + i*relu(g) ; h = o*relu(c)
//
// Block = one 16-col H-slice (blockIdx.x = hsl, 32 blocks). Its 8 waves all
// consume the SAME 4 gate-weight panels, so the block stages them in LDS:
// K-chunks of 128 (16KB panel), double-buffered (32KB), filled with
// global_load_async_to_lds_b128 running two stages ahead (ASYNCcnt).
// Wave w owns batch rows w*32..w*32+31 (two m-tiles): 8 WMMAs per 32-k step,
// B fragments from LDS (ds_load_b128), A fragments direct from L2.
// ---------------------------------------------------------------------------
__global__ void __launch_bounds__(256)
lstm_step_kernel(const unsigned short* __restrict__ xbf,   // [B*SEQ, EMBED]
                 const unsigned short* __restrict__ wut,   // [G4H, KTOT]
                 const float* __restrict__ bias,           // [G4H]
                 const unsigned short* __restrict__ hprev, // [B, HID] bf16
                 unsigned short* __restrict__ hnext,       // [B, HID] bf16
                 float* __restrict__ c,                    // [B, HID] f32
                 int t) {
    __shared__ __align__(16) unsigned char panel[2][PANEL_BYTES];

    const int tid  = threadIdx.x;
    const int lane = tid & 31;
    const int w    = tid >> 5;       // wave in block: 0..7
    const int hsl  = blockIdx.x;     // 0..31 H-slice
    const int m0   = w << 5;         // 32 batch rows per wave
    const int n0   = hsl << 4;
    const int kg   = lane >> 4;      // k-group (0/1)
    const int ml   = lane & 15;      // lane's M for A / N for B,C
    const int nlane = n0 + ml;

    const __bf16* X  = (const __bf16*)xbf;
    const __bf16* WU = (const __bf16*)wut;
    const __bf16* Hp = (const __bf16*)hprev;

    // ---- async panel copy: 256 lanes x 64B = 16KB per stage ----
    // LDS panel layout: row = gate*16 + n (64 rows), 256B per row (128 k bf16)
    const int prow  = tid >> 2;              // 0..63
    const int pgate = prow >> 4;
    const int pn    = prow & 15;
    const int pcol  = (tid & 3) * 32;        // element offset within row (64B)
    const __bf16* psrc_base =
        WU + (size_t)(pgate * HID + n0 + pn) * KTOT + pcol;

    auto copy_stage = [&](int kt, int buf) {
        const __bf16* src = psrc_base + kt;
        unsigned dst = lds_off(&panel[buf][prow * 256 + (tid & 3) * 64]);
#pragma unroll
        for (int j = 0; j < 4; ++j)
            async_copy_b128(dst + j * 16, src + j * 8);
    };

    // A rows for the two m-tiles; x part then h part (boundary 256 = 2*KCH)
    const __bf16* ax0 = X  + ((size_t)(m0 + ml) * SEQ + t) * EMBED;
    const __bf16* ax1 = X  + ((size_t)(m0 + 16 + ml) * SEQ + t) * EMBED;
    const __bf16* ah0 = Hp + (size_t)(m0 + ml) * HID;
    const __bf16* ah1 = Hp + (size_t)(m0 + 16 + ml) * HID;
    auto a_src0 = [&](int kt) -> const __bf16* {
        return (kt < EMBED) ? (ax0 + kt + kg * 8) : (ah0 + (kt - EMBED) + kg * 8);
    };
    auto a_src1 = [&](int kt) -> const __bf16* {
        return (kt < EMBED) ? (ax1 + kt + kg * 8) : (ah1 + (kt - EMBED) + kg * 8);
    };

    // Prologue: stages 0 and 1 in flight (8 async ops outstanding per wave)
    copy_stage(0, 0);
    copy_stage(KCH, 1);

    v8f acc[8] = {v8f{}, v8f{}, v8f{}, v8f{}, v8f{}, v8f{}, v8f{}, v8f{}};

#pragma unroll
    for (int s = 0; s < NSTAGE; ++s) {
        const int kt0 = s * KCH;
        wait_asynccnt_le4();      // stage s landed; stage s+1 still in flight
        __syncthreads();

        const unsigned char* pb = panel[s & 1];
#pragma unroll
        for (int u = 0; u < 4; ++u) {      // 4 x 32-k substeps per stage
            const int kt = kt0 + u * 32;
            v16bf a0 = load_a16(a_src0(kt));
            v16bf a1 = load_a16(a_src1(kt));
#pragma unroll
            for (int g = 0; g < 4; ++g) {
                const __bf16* bp = (const __bf16*)
                    (pb + (size_t)(g * 16 + ml) * 256 + u * 64 + kg * 16);
                v16bf bm = load_b16_lds(bp);
                acc[2 * g + 0] = WMMA_BF16(a0, bm, acc[2 * g + 0]);
                acc[2 * g + 1] = WMMA_BF16(a1, bm, acc[2 * g + 1]);
            }
        }

        __syncthreads();                   // all waves done reading this buffer
        if (s + 2 < NSTAGE) copy_stage(kt0 + 2 * KCH, s & 1);
    }

    // Gate activations + cell update. C/D frag: lane = N, VGPR r -> M = r+8*kg.
    const float bi = bias[nlane];
    const float bf_ = bias[HID + nlane];
    const float bg = bias[2 * HID + nlane];
    const float bo = bias[3 * HID + nlane];
    __bf16* Hn = (__bf16*)hnext;
#pragma unroll
    for (int tile = 0; tile < 2; ++tile) {
#pragma unroll
        for (int r = 0; r < 8; ++r) {
            const int m = m0 + tile * 16 + kg * 8 + r;
            const size_t idx = (size_t)m * HID + nlane;
            float ig = sigmoidf_dev(acc[0 + tile][r] + bi);
            float fg = sigmoidf_dev(acc[2 + tile][r] + bf_);
            float zg = acc[4 + tile][r] + bg;
            float gg = zg > 0.0f ? zg : 0.0f;         // activation='relu'
            float og = sigmoidf_dev(acc[6 + tile][r] + bo);
            float cn = fg * c[idx] + ig * gg;
            c[idx] = cn;
            float hv = og * (cn > 0.0f ? cn : 0.0f);  // h = o * relu(c)
            Hn[idx] = (__bf16)hv;
        }
    }
}

// ---------------------------------------------------------------------------
// Kernel 5: generic WMMA GEMM  D = act(A[M,K] @ Bt[N,K]^T + bias)
// One wave per 16x16 output tile, 2-stage pipelined K loop.
// ---------------------------------------------------------------------------
__global__ void __launch_bounds__(256)
wmma_gemm_kernel(const unsigned short* __restrict__ Abf, int lda,
                 const unsigned short* __restrict__ Btbf, int ldb,
                 const float* __restrict__ bias,
                 int K, int Ntiles,
                 unsigned short* __restrict__ outBf,
                 float* __restrict__ outF, int ldo, int relu) {
    const int lane = threadIdx.x & 31;
    const int wave = (blockIdx.x * blockDim.x + threadIdx.x) >> 5;
    const int mtile = wave / Ntiles;
    const int ntile = wave - mtile * Ntiles;
    const int m0 = mtile << 4;
    const int kg = lane >> 4;
    const int nl = lane & 15;

    const __bf16* A  = (const __bf16*)Abf;
    const __bf16* Bt = (const __bf16*)Btbf;
    const __bf16* arow = A  + (size_t)(m0 + nl) * lda;
    const __bf16* brow = Bt + (size_t)((ntile << 4) + nl) * ldb;

    v8f acc = {};
    v16bf aC = load_a16(arow + kg * 8);
    v16bf bC = load_a16(brow + kg * 16 - ((kg * 16) & 16) + ((kg * 16) & 16));
    bC = load_a16(brow + kg * 8);
    for (int kt = 0; kt < K; kt += 32) {
        const int kn = kt + 32;
        v16bf aN = {}, bN = {};
        if (kn < K) {
            aN = load_a16(arow + kn + kg * 8);
            bN = load_a16(brow + kn + kg * 8);
        }
        acc = WMMA_BF16(aC, bC, acc);
        aC = aN; bC = bN;
    }

    const int n = (ntile << 4) + nl;
    const float bb = bias[n];
#pragma unroll
    for (int r = 0; r < 8; ++r) {
        const int m = m0 + kg * 8 + r;
        float v = acc[r] + bb;
        if (relu) v = v > 0.0f ? v : 0.0f;
        if (outBf) ((__bf16*)outBf)[(size_t)m * ldo + n] = (__bf16)v;
        else       outF[(size_t)m * ldo + n] = v;
    }
}

// ---------------------------------------------------------------------------
// Kernel 6: row softmax over U2=64 columns. One wave32 per row.
// ---------------------------------------------------------------------------
__global__ void softmax_kernel(const float* __restrict__ logits,
                               float* __restrict__ out) {
    const int row  = (blockIdx.x * blockDim.x + threadIdx.x) >> 5;
    const int lane = threadIdx.x & 31;
    if (row >= BATCH) return;
    float v0 = logits[(size_t)row * U2 + lane];
    float v1 = logits[(size_t)row * U2 + 32 + lane];
    float mx = fmaxf(v0, v1);
#pragma unroll
    for (int o = 16; o > 0; o >>= 1) mx = fmaxf(mx, __shfl_xor(mx, o, 32));
    float e0 = __expf(v0 - mx), e1 = __expf(v1 - mx);
    float s = e0 + e1;
#pragma unroll
    for (int o = 16; o > 0; o >>= 1) s += __shfl_xor(s, o, 32);
    float inv = 1.0f / s;
    out[(size_t)row * U2 + lane]      = e0 * inv;
    out[(size_t)row * U2 + 32 + lane] = e1 * inv;
}

// ---------------------------------------------------------------------------
// Host-side orchestration
// ---------------------------------------------------------------------------
extern "C" void kernel_launch(void* const* d_in, const int* in_sizes, int n_in,
                              void* d_out, int out_size, void* d_ws, size_t ws_size,
                              hipStream_t stream) {
    (void)in_sizes; (void)n_in; (void)out_size; (void)ws_size;

    const int*   inputs = (const int*)  d_in[0];
    const float* emb    = (const float*)d_in[1];
    const float* W      = (const float*)d_in[2];
    const float* U      = (const float*)d_in[3];
    const float* b      = (const float*)d_in[4];
    const float* W2     = (const float*)d_in[5];
    const float* b2     = (const float*)d_in[6];
    const float* W3     = (const float*)d_in[7];
    const float* b3     = (const float*)d_in[8];
    float* out = (float*)d_out;

    char* ws = (char*)d_ws;
    size_t off = 0;
    auto alloc = [&](size_t bytes) -> char* {
        char* p = ws + off;
        off += (bytes + 255) & ~(size_t)255;
        return p;
    };
    unsigned short* xbf  = (unsigned short*)alloc((size_t)BATCH * SEQ * EMBED * 2);
    unsigned short* wut  = (unsigned short*)alloc((size_t)G4H * KTOT * 2);
    unsigned short* w2t  = (unsigned short*)alloc((size_t)U1 * HID * 2);
    unsigned short* w3t  = (unsigned short*)alloc((size_t)U2 * U1 * 2);
    unsigned short* h0   = (unsigned short*)alloc((size_t)BATCH * HID * 2);
    unsigned short* h1   = (unsigned short*)alloc((size_t)BATCH * HID * 2);
    float*          cbuf = (float*)        alloc((size_t)BATCH * HID * 4);
    unsigned short* ybf  = (unsigned short*)alloc((size_t)BATCH * U1 * 2);
    float*          logz = (float*)        alloc((size_t)BATCH * U2 * 4);

    // 1) embedding gather -> bf16
    {
        int total = BATCH * SEQ * EMBED;
        embed_cast_kernel<<<total / 256, 256, 0, stream>>>(inputs, emb, xbf, total);
    }
    // 2) weight transposes -> bf16 [N,K] layout (WUt = [W;U]^T fused)
    transpose_cast_kernel<<<(EMBED * G4H + 255) / 256, 256, 0, stream>>>(
        W, wut, EMBED, G4H, KTOT, 0, EMBED * G4H);
    transpose_cast_kernel<<<(HID * G4H + 255) / 256, 256, 0, stream>>>(
        U, wut, HID, G4H, KTOT, EMBED, HID * G4H);
    transpose_cast_kernel<<<(HID * U1 + 255) / 256, 256, 0, stream>>>(
        W2, w2t, HID, U1, HID, 0, HID * U1);
    transpose_cast_kernel<<<(U1 * U2 + 255) / 256, 256, 0, stream>>>(
        W3, w3t, U1, U2, U1, 0, U1 * U2);
    // 3) zero h0, c
    init_state_kernel<<<(BATCH * HID + 255) / 256, 256, 0, stream>>>(
        h0, cbuf, BATCH * HID);

    // 4) 128 fused recurrent steps (double-buffered h; 32 blocks = 32 H-slices)
    unsigned short* hbuf[2] = {h0, h1};
    for (int t = 0; t < SEQ; ++t) {
        lstm_step_kernel<<<32, 256, 0, stream>>>(
            xbf, wut, b, hbuf[t & 1], hbuf[(t + 1) & 1], cbuf, t);
    }
    unsigned short* hfin = hbuf[SEQ & 1];  // SEQ even -> h0

    // 5) y = relu(h @ W2 + b2) -> bf16 [256, 256]
    wmma_gemm_kernel<<<32, 256, 0, stream>>>(
        hfin, HID, w2t, HID, b2, HID, U1 / 16, ybf, nullptr, U1, 1);
    // 6) logits = y @ W3 + b3 -> f32 [256, 64]
    wmma_gemm_kernel<<<8, 256, 0, stream>>>(
        ybf, U1, w3t, U1, b3, U1, U2 / 16, nullptr, logz, U2, 0);
    // 7) row softmax -> d_out
    softmax_kernel<<<32, 256, 0, stream>>>(logz, out);
}
